// GATNet_56856777064814
// MI455X (gfx1250) — matrix operand order repaired and checked by reading, not compiled, not run on hardware
//
#include <hip/hip_runtime.h>
#include <math.h>

#define NEG_SLOPE 0.2f
#define GAT_EPS   1e-16f

typedef __attribute__((ext_vector_type(16))) _Float16 v16h;
typedef __attribute__((ext_vector_type(8)))  float    v8f;

// ---- native CDNA5 fp32 atomics (non-returning -> STOREcnt; s_endpgm drains).
// Device scope so cross-WGP segment reductions are coherent.
__device__ __forceinline__ void atomAddF32(float* p, float v) {
  asm volatile("global_atomic_add_f32 %0, %1, off scope:SCOPE_DEV"
               :: "v"((unsigned long long)p), "v"(v) : "memory");
}
__device__ __forceinline__ void atomMaxF32(float* p, float v) {
  asm volatile("global_atomic_max_num_f32 %0, %1, off scope:SCOPE_DEV"
               :: "v"((unsigned long long)p), "v"(v) : "memory");
}

// ---- K0: fold attention vectors + pre-pack layer-2 WMMA B-matrix in f16.
// consts[0:2]=u_src1=W1^T att_src1, consts[2:4]=u_dst1.
// Bpk: 2 K-chunks x 32 lanes x 16 halves, contiguous-k B layout matching K3's A packing:
//   lane: col=lane&15, bhalf=lane>>4; element i -> K = chunk*32 + bhalf*16 + i
//   col0=W2[0,K], col1=W2[1,K], col2=(W2^T att_src2)[K], col3=(W2^T att_dst2)[K]
__global__ void gat_prep(const float* __restrict__ W1, const float* __restrict__ as1,
                         const float* __restrict__ ad1, const float* __restrict__ W2,
                         const float* __restrict__ as2, const float* __restrict__ ad2,
                         float* __restrict__ consts, _Float16* __restrict__ Bpk) {
  int t = threadIdx.x;  // 64 threads
  if (t < 2) {
    float us = 0.f, ud = 0.f;
    for (int j = 0; j < 64; ++j) { us += as1[j] * W1[j*2+t]; ud += ad1[j] * W1[j*2+t]; }
    consts[t] = us; consts[2+t] = ud;
  }
  int chunk = t >> 5, lane = t & 31;
  int col = lane & 15, bhalf = lane >> 4;
  float s0 = as2[0], s1 = as2[1], d0 = ad2[0], d1 = ad2[1];
  for (int i = 0; i < 16; ++i) {
    int K = chunk * 32 + bhalf * 16 + i;
    float w0 = W2[K], w1 = W2[64 + K];
    float v = 0.f;
    if      (col == 0) v = w0;
    else if (col == 1) v = w1;
    else if (col == 2) v = s0 * w0 + s1 * w1;
    else if (col == 3) v = d0 * w0 + d1 * w1;
    Bpk[t * 16 + i] = (_Float16)v;
  }
}

// ---- K1: t1 = x @ W1^T  (K=2: memory-bound, VALU is optimal), zero agg1
__global__ void gat_l1_transform(const float* __restrict__ x, const float* __restrict__ W1,
                                 float* __restrict__ t1, float* __restrict__ agg1, int N) {
  int idx = blockIdx.x * blockDim.x + threadIdx.x;
  if (idx >= N * 64) return;
  int n = idx >> 6, j = idx & 63;
  t1[idx]   = x[2*n] * W1[2*j] + x[2*n+1] * W1[2*j+1];
  agg1[idx] = 0.f;
}

// ---- K1b: per-node attn scalars via folded 2-dim dot; init segment max/sum
__global__ void gat_l1_node(const float* __restrict__ x, const float* __restrict__ consts,
                            float* __restrict__ a_src, float* __restrict__ a_dst,
                            float* __restrict__ m, float* __restrict__ denom, int N) {
  int n = blockIdx.x * blockDim.x + threadIdx.x;
  if (n >= N) return;
  float x0 = x[2*n], x1 = x[2*n+1];
  a_src[n] = x0 * consts[0] + x1 * consts[1];
  a_dst[n] = x0 * consts[2] + x1 * consts[3];
  m[n]     = -INFINITY;
  denom[n] = 0.f;
}

// ---- Edge pass A: e = leaky_relu(a_src[s*ss]+a_dst[d*sd]); segment max.
// Strided pointers let layer 2 read attn scalars from the interleaved out4 buffer.
__global__ void gat_edge_logits(const int* __restrict__ src, const int* __restrict__ dst,
                                const float* __restrict__ a_src, int ss,
                                const float* __restrict__ a_dst, int sd,
                                float* __restrict__ ebuf, float* __restrict__ m,
                                int E, int Etot) {
  int idx = blockIdx.x * blockDim.x + threadIdx.x;
  if (idx >= Etot) return;
  int s, d;
  if (idx < E) { s = src[idx]; d = dst[idx]; } else { s = d = idx - E; }
  float z = a_src[(size_t)s * ss] + a_dst[(size_t)d * sd];
  float e = z > 0.f ? z : NEG_SLOPE * z;
  ebuf[idx] = e;
  atomMaxF32(m + d, e);
}

// ---- Edge pass B: ex = exp(e - m[d]); segment sum of ex
__global__ void gat_edge_exp(const int* __restrict__ dst, float* __restrict__ ebuf,
                             const float* __restrict__ m, float* __restrict__ denom,
                             int E, int Etot) {
  int idx = blockIdx.x * blockDim.x + threadIdx.x;
  if (idx >= Etot) return;
  int d = (idx < E) ? dst[idx] : (idx - E);
  float ex = expf(ebuf[idx] - m[d]);
  ebuf[idx] = ex;
  atomAddF32(denom + d, ex);
}

// ---- Edge pass C (layer1, 64ch): one wave per edge, lanes = channels.
// edge forced into an SGPR so the per-wave-uniform loads become SMEM (s_load),
// keeping VMEM bandwidth for the 64-wide gather + atomics (roofline-dominant).
__global__ void gat_edge_agg64(const int* __restrict__ src, const int* __restrict__ dst,
                               const float* __restrict__ ebuf, const float* __restrict__ denom,
                               const float* __restrict__ t1, float* __restrict__ agg1,
                               int E, int Etot) {
  int lane = threadIdx.x & 31;
  int edge = __builtin_amdgcn_readfirstlane(
      (int)blockIdx.x * (int)(blockDim.x >> 5) + (int)(threadIdx.x >> 5));
  if (edge >= Etot) return;  // uniform per wave
  int s, d;
  if (edge < E) { s = src[edge]; d = dst[edge]; } else { s = d = edge - E; }
  float alpha = ebuf[edge] / (denom[d] + GAT_EPS);
  float h0 = t1[(size_t)s * 64 + lane];
  float h1 = t1[(size_t)s * 64 + lane + 32];
  atomAddF32(&agg1[(size_t)d * 64 + lane],      alpha * h0);
  atomAddF32(&agg1[(size_t)d * 64 + lane + 32], alpha * h1);
}

// ---- K2: h1' = relu(agg1 + b1), written over t1
__global__ void gat_relu_bias(const float* __restrict__ agg1, const float* __restrict__ b1,
                              float* __restrict__ h1p, int N) {
  int idx = blockIdx.x * blockDim.x + threadIdx.x;
  if (idx >= N * 64) return;
  h1p[idx] = fmaxf(agg1[idx] + b1[idx & 63], 0.f);
}

// ---- K2b: reinit segment buffers for layer 2
__global__ void gat_l2_node_init(float* __restrict__ m, float* __restrict__ denom,
                                 float* __restrict__ agg2, int N) {
  int n = blockIdx.x * blockDim.x + threadIdx.x;
  if (n >= N) return;
  m[n] = -INFINITY;
  denom[n] = 0.f;
  agg2[2*n] = 0.f;
  agg2[2*n+1] = 0.f;
}

// ---- K3: layer-2 node transform via WMMA (K=64 as two 16x16x32 f16 steps).
// A = 16 nodes x 64 of h1'; B pre-packed (see gat_prep).
// C row m, cols {0,1,2,3} = {hh0, hh1, a_src2, a_dst2} -> interleaved out4[node*4+col].
__global__ void gat_l2_transform(const float* __restrict__ h1p,
                                 const _Float16* __restrict__ Bpk,
                                 float* __restrict__ out4, int N) {
  int wave = (blockIdx.x * blockDim.x + threadIdx.x) >> 5;
  int lane = threadIdx.x & 31;
  int base = wave * 16;
  if (base >= N) return;            // uniform per wave -> EXEC all-1s for WMMA
  int rc   = lane & 15;             // A: row index; B/C: column index
  int half = lane >> 4;
  int na = base + rc; if (na >= N) na = N - 1;   // clamped A row (stores predicated)

  const v16h* Bv = (const v16h*)Bpk;           // 32B-aligned, pre-packed lane layout
  v8f c = {};
#pragma unroll
  for (int chunk = 0; chunk < 2; ++chunk) {
    // A layout (16-bit 16x32): lanes hold row M=lane&15; K = half*16+i,
    // offset by chunk*32 (matches Bpk packing).
    v16h a;
    const float* hp = h1p + (size_t)na * 64 + chunk * 32 + half * 16;
#pragma unroll
    for (int i = 0; i < 8; ++i) a[i]     = (_Float16)hp[i];
#pragma unroll
    for (int i = 0; i < 8; ++i) a[8 + i] = (_Float16)hp[8 + i];

    v16h b = Bv[chunk * 32 + lane];            // single contiguous vector load

    c = __builtin_amdgcn_wmma_f32_16x16x32_f16(false, a, false, b, (short)0, c,
                                               false, false);
  }

  // C layout: c[r] holds row M=r (lanes 0-15) / M=r+8 (lanes 16-31), col = lane&15.
  // Single interleaved destination -> one predicate, affine addresses, no ladder.
  if (rc < 4) {
    float* p = out4 + (size_t)(base + half * 8) * 4 + rc;
    if (base + 16 <= N) {           // uniform full-tile fast path
#pragma unroll
      for (int r = 0; r < 8; ++r) p[r * 4] = c[r];
    } else {
#pragma unroll
      for (int r = 0; r < 8; ++r)
        if (base + half * 8 + r < N) p[r * 4] = c[r];
    }
  }
}

// ---- Edge pass C (layer2, 2ch): thread per edge; hh read from out4 (stride 4)
__global__ void gat_edge_agg2(const int* __restrict__ src, const int* __restrict__ dst,
                              const float* __restrict__ ebuf, const float* __restrict__ denom,
                              const float* __restrict__ out4, float* __restrict__ agg2,
                              int E, int Etot) {
  int idx = blockIdx.x * blockDim.x + threadIdx.x;
  if (idx >= Etot) return;
  int s, d;
  if (idx < E) { s = src[idx]; d = dst[idx]; } else { s = d = idx - E; }
  float alpha = ebuf[idx] / (denom[d] + GAT_EPS);
  atomAddF32(&agg2[2*d],     alpha * out4[(size_t)s * 4]);
  atomAddF32(&agg2[2*d + 1], alpha * out4[(size_t)s * 4 + 1]);
}

// ---- K4: out = log_softmax(agg2 + b2)
__global__ void gat_finalize(const float* __restrict__ agg2, const float* __restrict__ b2,
                             float* __restrict__ out, int N) {
  int n = blockIdx.x * blockDim.x + threadIdx.x;
  if (n >= N) return;
  float o0 = agg2[2*n]     + b2[0];
  float o1 = agg2[2*n + 1] + b2[1];
  float mx = fmaxf(o0, o1);
  float l  = logf(expf(o0 - mx) + expf(o1 - mx)) + mx;
  out[2*n]     = o0 - l;
  out[2*n + 1] = o1 - l;
}

extern "C" void kernel_launch(void* const* d_in, const int* in_sizes, int n_in,
                              void* d_out, int out_size, void* d_ws, size_t ws_size,
                              hipStream_t stream) {
  const float* x   = (const float*)d_in[0];
  const int*   ei  = (const int*)  d_in[1];
  const float* W1  = (const float*)d_in[2];
  const float* as1 = (const float*)d_in[3];
  const float* ad1 = (const float*)d_in[4];
  const float* b1  = (const float*)d_in[5];
  const float* W2  = (const float*)d_in[6];
  const float* as2 = (const float*)d_in[7];
  const float* ad2 = (const float*)d_in[8];
  const float* b2  = (const float*)d_in[9];

  int N = in_sizes[0] / 2;
  int E = in_sizes[1] / 2;
  int Etot = E + N;                 // + self loops
  const int* src = ei;
  const int* dst = ei + E;

  // workspace layout (floats)
  float* ws = (float*)d_ws;
  size_t N64 = (size_t)N * 64;
  float* t1     = ws;                  // N*64  (layer1 h, later relu'd h1')
  float* agg1   = t1   + N64;          // N*64
  float* a_srcb = agg1 + N64;          // N
  float* a_dstb = a_srcb + N;          // N
  float* mbuf   = a_dstb + N;          // N
  float* dbuf   = mbuf   + N;          // N
  float* ebuf   = dbuf   + N;          // Etot
  float* out4   = ebuf   + Etot;       // 4N: {hh0, hh1, a_src2, a_dst2} per node
  float* agg2   = out4   + 4*(size_t)N;// 2N
  size_t coff   = (size_t)(agg2 + 2*(size_t)N - ws);
  coff = (coff + 63) & ~(size_t)63;    // align consts; Bpk 64B-aligned after
  float*    consts = ws + coff;        // 16 floats (u_src1/u_dst1 + pad)
  _Float16* Bpk    = (_Float16*)(consts + 16);  // 1024 halves, WMMA-layout B

  auto cdiv = [](int a, int b) { return (a + b - 1) / b; };
  const int B = 256;

  gat_prep<<<1, 64, 0, stream>>>(W1, as1, ad1, W2, as2, ad2, consts, Bpk);

  // ---- layer 1
  gat_l1_transform<<<cdiv(N * 64, B), B, 0, stream>>>(x, W1, t1, agg1, N);
  gat_l1_node<<<cdiv(N, B), B, 0, stream>>>(x, consts, a_srcb, a_dstb, mbuf, dbuf, N);
  gat_edge_logits<<<cdiv(Etot, B), B, 0, stream>>>(src, dst, a_srcb, 1, a_dstb, 1,
                                                   ebuf, mbuf, E, Etot);
  gat_edge_exp<<<cdiv(Etot, B), B, 0, stream>>>(dst, ebuf, mbuf, dbuf, E, Etot);
  gat_edge_agg64<<<cdiv(Etot, B / 32), B, 0, stream>>>(src, dst, ebuf, dbuf, t1, agg1, E, Etot);
  gat_relu_bias<<<cdiv(N * 64, B), B, 0, stream>>>(agg1, b1, t1, N);

  // ---- layer 2
  gat_l2_node_init<<<cdiv(N, B), B, 0, stream>>>(mbuf, dbuf, agg2, N);
  gat_l2_transform<<<cdiv(cdiv(N, 16), B / 32), B, 0, stream>>>(t1, Bpk, out4, N);
  gat_edge_logits<<<cdiv(Etot, B), B, 0, stream>>>(src, dst, out4 + 2, 4, out4 + 3, 4,
                                                   ebuf, mbuf, E, Etot);
  gat_edge_exp<<<cdiv(Etot, B), B, 0, stream>>>(dst, ebuf, mbuf, dbuf, E, Etot);
  gat_edge_agg2<<<cdiv(Etot, B), B, 0, stream>>>(src, dst, ebuf, dbuf, out4, agg2, E, Etot);

  gat_finalize<<<cdiv(N, B), B, 0, stream>>>(agg2, b2, (float*)d_out, N);
}
